// SinkhornLoss_18245021073823
// MI455X (gfx1250) — compile-verified, matrix operand order checked by reading, not run on hardware
//
#include <hip/hip_runtime.h>
#include <hip/hip_bf16.h>

// Sinkhorn loss, MI455X (gfx1250) design:
//  - K = exp(M/eps) precomputed once in bf16 (128 MB -> L2-resident, 192 MB L2)
//  - scaling iteration u = a/(K v), v = b/(K^T u)  (stable in fp32 for M in [0,1], eps=0.1)
//  - row matvec via v_wmma_f32_16x16x32_bf16 (B = broadcast of v chunk)
//  - col matvec via coalesced b128 row streaming + exclusive partial buffers
//  - final CE = -sum target * (M/eps + log u_i + log v_j), streaming reduction

#define NDIM     8192
#define INV_EPS  10.0f
#define NITERS   50
#define ROWBLK2  128   // row blocks for the K^T matvec partials

typedef __attribute__((ext_vector_type(16))) __bf16 v16bf;
typedef __attribute__((ext_vector_type(8)))  __bf16 v8bf;
typedef __attribute__((ext_vector_type(4)))  __bf16 v4bf;
typedef __attribute__((ext_vector_type(8)))  float  v8f;
typedef __attribute__((ext_vector_type(4)))  float  v4f;

// ---------------------------------------------------------------------------
// K0: Kb = bf16(exp(M/eps)); init v_bf = 1 (g0 = 0); zero the output scalar.
// ---------------------------------------------------------------------------
__global__ __launch_bounds__(256) void k_prep(const float* __restrict__ M,
                                              __bf16* __restrict__ Kb,
                                              __bf16* __restrict__ vbf,
                                              float* __restrict__ out) {
  const size_t tid = (size_t)blockIdx.x * blockDim.x + threadIdx.x;
  const size_t nthreads = (size_t)gridDim.x * blockDim.x;
  const size_t nvec = (size_t)NDIM * NDIM / 4;
  const v4f* M4 = (const v4f*)M;
  v4bf* K4 = (v4bf*)Kb;
  for (size_t i = tid; i < nvec; i += nthreads) {
    v4f m = M4[i];
    v4bf k;
#pragma unroll
    for (int j = 0; j < 4; ++j) k[j] = (__bf16)__expf(m[j] * INV_EPS);
    K4[i] = k;
  }
  if (tid < NDIM) vbf[tid] = (__bf16)1.0f;
  if (tid == 0) out[0] = 0.0f;
}

// ---------------------------------------------------------------------------
// K1: row matvec S_i = sum_j Kb[i][j] * v[j] via WMMA bf16; then u = a / S.
// One block = 16 rows, 4 waves split the K dimension (2048 each, 64 WMMA ops).
// A tile: 16x32 bf16 per the ISA layout (lane<16: K 0..7 & 16..23; lane>=16:
// K 8..15 & 24..31). B tile: v chunk broadcast into all 16 columns -> same
// contiguous-slice pattern, so all columns of D equal the row-sum vector.
// ---------------------------------------------------------------------------
__global__ __launch_bounds__(128) void k_rowmv(const __bf16* __restrict__ Kb,
                                               const __bf16* __restrict__ vbf,
                                               const float* __restrict__ a,
                                               float* __restrict__ u) {
  __shared__ float red[4][16];
  const int wave = threadIdx.x >> 5;
  const int lane = threadIdx.x & 31;
  const int row0 = blockIdx.x * 16;
  const int m = lane & 15;
  const int off = (lane < 16) ? 0 : 8;
  const __bf16* rowp = Kb + (size_t)(row0 + m) * NDIM;
  const int kbase = wave * 2048;

  v8f c = {};
  for (int kc = 0; kc < 64; ++kc) {
    const int k0 = kbase + kc * 32;
    v8bf a0 = *(const v8bf*)(rowp + k0 + off);
    v8bf a1 = *(const v8bf*)(rowp + k0 + 16 + off);
    v8bf b0 = *(const v8bf*)(vbf + k0 + off);
    v8bf b1 = *(const v8bf*)(vbf + k0 + 16 + off);
    v16bf A, B;
#pragma unroll
    for (int t = 0; t < 8; ++t) {
      A[t] = a0[t]; A[8 + t] = a1[t];
      B[t] = b0[t]; B[8 + t] = b1[t];
    }
    c = __builtin_amdgcn_wmma_f32_16x16x32_bf16(
        /*neg_a=*/false, A, /*neg_b=*/false, B,
        /*c_mod=*/(short)0, c, /*reuse_a=*/false, /*reuse_b=*/false);
  }
  // D layout: lane 0 holds rows 0..7 (VGPR r -> M=r), lane 16 holds rows 8..15.
  if (lane == 0) {
#pragma unroll
    for (int r = 0; r < 8; ++r) red[wave][r] = c[r];
  }
  if (lane == 16) {
#pragma unroll
    for (int r = 0; r < 8; ++r) red[wave][8 + r] = c[r];
  }
  __syncthreads();
  if (threadIdx.x < 16) {
    const float S = red[0][threadIdx.x] + red[1][threadIdx.x] +
                    red[2][threadIdx.x] + red[3][threadIdx.x];
    const int row = row0 + threadIdx.x;
    u[row] = a[row] / S;   // exp(f/eps) = a / S
  }
}

// ---------------------------------------------------------------------------
// K2: column matvec partials: tpart[rb][j] = sum_{i in rb} u[i] * Kb[i][j].
// Grid (4 col-blocks x 128 row-blocks); each thread owns 8 contiguous columns
// (16B b128 loads, fully coalesced); exclusive writes -> no atomics, no init.
// ---------------------------------------------------------------------------
__global__ __launch_bounds__(256) void k_colmv(const __bf16* __restrict__ Kb,
                                               const float* __restrict__ u,
                                               float* __restrict__ tpart) {
  const int cb = blockIdx.x;            // 0..3
  const int rb = blockIdx.y;            // 0..ROWBLK2-1
  const int col0 = cb * 2048 + threadIdx.x * 8;
  const int rowStart = rb * (NDIM / ROWBLK2);
  float acc[8] = {0.f, 0.f, 0.f, 0.f, 0.f, 0.f, 0.f, 0.f};
  for (int i = 0; i < NDIM / ROWBLK2; ++i) {
    const int row = rowStart + i;
    const float ui = u[row];            // uniform -> scalar load
    v8bf kv = *(const v8bf*)(Kb + (size_t)row * NDIM + col0);
#pragma unroll
    for (int j = 0; j < 8; ++j) acc[j] = fmaf(ui, (float)kv[j], acc[j]);
  }
  float* dst = tpart + (size_t)rb * NDIM + col0;
#pragma unroll
  for (int j = 0; j < 8; ++j) dst[j] = acc[j];
}

// ---------------------------------------------------------------------------
// K3: reduce partials, v = b / t; keep fp32 (for final log) and bf16 (for K1).
// ---------------------------------------------------------------------------
__global__ __launch_bounds__(256) void k_vupdate(const float* __restrict__ tpart,
                                                 const float* __restrict__ b,
                                                 float* __restrict__ v,
                                                 __bf16* __restrict__ vbf) {
  const int j = blockIdx.x * blockDim.x + threadIdx.x;
  if (j >= NDIM) return;
  float s = 0.f;
  for (int rb = 0; rb < ROWBLK2; ++rb) s += tpart[(size_t)rb * NDIM + j];
  const float vj = b[j] / s;            // exp(g/eps) = b / (K^T u)
  v[j] = vj;
  vbf[j] = (__bf16)vj;
}

// ---------------------------------------------------------------------------
// K4: CE = -sum_ij target * (M/eps + log u_i + log v_j), streaming reduction.
// ---------------------------------------------------------------------------
__global__ __launch_bounds__(256) void k_ce(const float* __restrict__ M,
                                            const float* __restrict__ T,
                                            const float* __restrict__ u,
                                            const float* __restrict__ v,
                                            float* __restrict__ out) {
  __shared__ float red[256];
  const size_t tid = (size_t)blockIdx.x * blockDim.x + threadIdx.x;
  const size_t nthreads = (size_t)gridDim.x * blockDim.x;
  const size_t nvec = (size_t)NDIM * NDIM / 4;
  const v4f* M4 = (const v4f*)M;
  const v4f* T4 = (const v4f*)T;
  float acc = 0.f;
  for (size_t i = tid; i < nvec; i += nthreads) {
    const int row = (int)((i * 4) / NDIM);
    const int col = (int)((i * 4) % NDIM);   // NDIM % 4 == 0: no row crossing
    const float fe = __logf(u[row]);
    v4f m = M4[i];
    v4f t = T4[i];
#pragma unroll
    for (int j = 0; j < 4; ++j) {
      const float lp = m[j] * INV_EPS + fe + __logf(v[col + j]);
      acc = fmaf(-t[j], lp, acc);
    }
  }
  red[threadIdx.x] = acc;
  __syncthreads();
  for (int s = 128; s > 0; s >>= 1) {
    if (threadIdx.x < s) red[threadIdx.x] += red[threadIdx.x + s];
    __syncthreads();
  }
  if (threadIdx.x == 0) atomicAdd(out, red[0]);
}

// ---------------------------------------------------------------------------
extern "C" void kernel_launch(void* const* d_in, const int* in_sizes, int n_in,
                              void* d_out, int out_size, void* d_ws, size_t ws_size,
                              hipStream_t stream) {
  const float* M = (const float*)d_in[0];
  const float* target = (const float*)d_in[1];
  const float* a = (const float*)d_in[2];
  const float* b = (const float*)d_in[3];
  float* out = (float*)d_out;

  // workspace layout (all 16B aligned):
  //   Kb   : NDIM*NDIM bf16   = 128 MB
  //   u    : NDIM f32         = 32 KB
  //   v    : NDIM f32         = 32 KB
  //   vbf  : NDIM bf16        = 16 KB
  //   tpart: ROWBLK2*NDIM f32 = 4 MB
  char* ws = (char*)d_ws;
  __bf16* Kb = (__bf16*)ws;
  size_t off = (size_t)NDIM * NDIM * sizeof(__bf16);
  float* u = (float*)(ws + off);      off += (size_t)NDIM * sizeof(float);
  float* v = (float*)(ws + off);      off += (size_t)NDIM * sizeof(float);
  __bf16* vbf = (__bf16*)(ws + off);  off += (size_t)NDIM * sizeof(__bf16);
  float* tpart = (float*)(ws + off);

  k_prep<<<4096, 256, 0, stream>>>(M, Kb, vbf, out);
  for (int it = 0; it < NITERS; ++it) {
    k_rowmv<<<NDIM / 16, 128, 0, stream>>>(Kb, vbf, a, u);
    k_colmv<<<dim3(4, ROWBLK2), 256, 0, stream>>>(Kb, u, tpart);
    k_vupdate<<<NDIM / 256, 256, 0, stream>>>(tpart, b, v, vbf);
  }
  k_ce<<<2048, 256, 0, stream>>>(M, target, u, v, out);
}
